// BiLSTM_CRF_19215683682697
// MI455X (gfx1250) — matrix operand order repaired and checked by reading, not compiled, not run on hardware
//
#include <hip/hip_runtime.h>
#include <hip/hip_bf16.h>
#include <math.h>

// ---------------- problem constants ----------------
#define BB   32          // batch
#define SS   256         // seq len
#define EE   300         // embed dim
#define EP   320         // embed dim padded to K-multiple of 32
#define HH   512         // hidden
#define G4   2048        // 4*H
#define TT   36          // tags
#define MM   (SS*BB)     // 8192 rows of the big GEMMs
#define START_TAG 34
#define END_TAG   35
#define NEGV  (-10000.0f)

typedef __bf16 bf16;
typedef __attribute__((ext_vector_type(16))) __bf16 v16bf;
typedef __attribute__((ext_vector_type(8)))  __bf16 v8bf;
typedef __attribute__((ext_vector_type(8)))  float  v8f;

// ---------------- workspace layout (bytes, 256-aligned) ----------------
#define OFF_X     0                        // [MM][EP] bf16              = 5,242,880
#define OFF_WIH   5242880ull               // [2][G4][EP] bf16           = 2,621,440
#define OFF_WHH   7864320ull               // [2][G4][HH] bf16           = 4,194,304
#define OFF_XP    12058624ull              // [2][MM][G4] f32            = 134,217,728
#define OFF_HS    146276352ull             // [2][MM][HH] f32            = 33,554,432
#define OFF_FEATS 179830784ull             // [MM][TT] f32               = 1,179,648
#define OFF_BPS   181010432ull             // [BB][SS][TT] i32           = 1,179,648

// ---------------- WMMA helpers (CDNA5 16x16x32 bf16) ----------------
// A-matrix 16x32 bf16 per-lane layout (contiguous, matches B):
//   lanes 0-15 : row M=lane, K = 0..15
//   lanes 16-31: row M=lane-16, K = 16..31
__device__ __forceinline__ v16bf load_a_frag(const bf16* row, int kk, int lane) {
    return *(const v16bf*)(row + kk + ((lane < 16) ? 0 : 16));
}

// B-matrix 32x16 bf16: lane n%16 = column, lanes 0-15 hold K=0..15,
// lanes 16-31 hold K=16..31 -> with W stored [N][K] row-major this is one
// contiguous 32-byte load per lane.
__device__ __forceinline__ v16bf load_b_frag(const bf16* wrow, int kk, int lane) {
    return *(const v16bf*)(wrow + kk + ((lane < 16) ? 0 : 16));
}

__device__ __forceinline__ v8f wmma_bf16(v16bf a, v16bf b, v8f c) {
    // (neg_a, A, neg_b, B, c_mod, C, reuse_a, reuse_b)
    return __builtin_amdgcn_wmma_f32_16x16x32_bf16(false, a, false, b, (short)0, c,
                                                   false, false);
}

__device__ __forceinline__ float sigf(float x) {
    return 1.0f / (1.0f + __expf(-x));
}

// ---------------- prep kernels ----------------
__global__ void cast_pad_kernel(const float* __restrict__ src, bf16* __restrict__ dst,
                                int rows, int cols, int pitch) {
    int idx = blockIdx.x * blockDim.x + threadIdx.x;
    int total = rows * pitch;
    if (idx >= total) return;
    int r = idx / pitch, k = idx - r * pitch;
    dst[idx] = (k < cols) ? (bf16)src[(size_t)r * cols + k] : (bf16)0.0f;
}

__global__ void embed_kernel(const int* __restrict__ sent, const float* __restrict__ emb,
                             bf16* __restrict__ xbf) {
    int idx = blockIdx.x * blockDim.x + threadIdx.x;   // over MM*EP
    if (idx >= MM * EP) return;
    int m = idx / EP, k = idx - m * EP;
    int s = m / BB, b = m - s * BB;
    int tok = sent[b * SS + s];                        // sentences is [B,S]
    xbf[idx] = (k < EE) ? (bf16)emb[(size_t)tok * EE + k] : (bf16)0.0f;
}

// ---------------- input projection: xp = x @ W_ih^T + bias ----------------
// grid (4, 512, 2), block 256 (8 waves); each wave owns 4 N-tiles of one
// M-tile row. K-outer loop: A fragment loaded once per k-chunk, feeds 4
// independent WMMA accumulators (no D->C serial chain).
__global__ void __launch_bounds__(256)
ih_gemm_kernel(const bf16* __restrict__ xbf,   // [MM][EP]
               const bf16* __restrict__ Wih,   // [2][G4][EP]
               const float* __restrict__ b_f, const float* __restrict__ b_b,
               float* __restrict__ xp)         // [2][MM][G4]
{
    const int tid = threadIdx.x, lane = tid & 31, wave = tid >> 5;
    const int ntb = (blockIdx.x * 8 + wave) * 4;   // first of 4 N-tiles, 0..124
    const int mtile = blockIdx.y;                  // 0..511
    const int dir = blockIdx.z;
    const bf16*  W    = Wih + (size_t)dir * G4 * EP;
    const float* bias = dir ? b_b : b_f;
    float*       out  = xp + (size_t)dir * MM * G4;

    const int lcol = lane & 15;
    const bf16* arow = xbf + (size_t)(mtile * 16 + lcol) * EP;

    v8f acc[4] = {};
#pragma unroll 2
    for (int kk = 0; kk < EP; kk += 32) {
        v16bf a = load_a_frag(arow, kk, lane);
#pragma unroll
        for (int jt = 0; jt < 4; ++jt) {
            const bf16* wrow = W + (size_t)((ntb + jt) * 16 + lcol) * EP;
            v16bf b = load_b_frag(wrow, kk, lane);
            acc[jt] = wmma_bf16(a, b, acc[jt]);
        }
    }
    const int m0 = mtile * 16 + ((lane < 16) ? 0 : 8);
#pragma unroll
    for (int jt = 0; jt < 4; ++jt) {
        const int ncol = (ntb + jt) * 16 + lcol;
        const float bb = bias[ncol];
#pragma unroll
        for (int i = 0; i < 8; ++i)
            out[(size_t)(m0 + i) * G4 + ncol] = acc[i == i ? jt : jt][i] + bb;
    }
}

// ---------------- persistent recurrent kernel ----------------
// grid 2 (one block per direction), 1024 threads = 32 waves on one WGP.
// dynamic LDS: g_pre f32 [32][2048] (256KB) + h bf16 [32][512] (32KB) = 288KB
// -> only legal on CDNA5's 320KB-per-WGP LDS.
// Per step: M=32, N=2048, K=512.  Wave w -> M-tile (w&1), N-tiles
// (w>>1)*8 .. +7.  K-outer loop: one LDS A-fragment load feeds 8 independent
// WMMA accumulators, so the serial recurrence's critical path has no
// WMMA->WMMA RAW hazards and 8x less LDS read traffic.
__global__ void __launch_bounds__(1024)
lstm_kernel(const float* __restrict__ xp,   // [2][MM][G4]
            const bf16*  __restrict__ Whh,  // [2][G4][HH]
            const float* __restrict__ h0,   // [2][B][H]
            const float* __restrict__ c0,   // [2][B][H]
            float* __restrict__ hs)         // [2][MM][HH]
{
    extern __shared__ char smem[];
    float* g_pre = (float*)smem;                        // [32][2048]
    bf16*  h_lds = (bf16*)(smem + (size_t)BB * G4 * 4); // [32][512]

    const int dir  = blockIdx.x;
    const int tid  = threadIdx.x;
    const int lane = tid & 31, wave = tid >> 5;
    const float* xp_d = xp + (size_t)dir * MM * G4;
    const bf16*  W    = Whh + (size_t)dir * G4 * HH;
    float*       hs_d = hs + (size_t)dir * MM * HH;

    // init hidden/cell state: 16384 (b,j) pairs, 16 per thread, c in registers
    float c_reg[16];
#pragma unroll
    for (int q = 0; q < 16; ++q) {
        int p = q * 1024 + tid;
        c_reg[q] = c0[dir * BB * HH + p];
        h_lds[p] = (bf16)h0[dir * BB * HH + p];
    }
    __syncthreads();

    const int mt   = wave & 1;          // M tile (batch rows 0-15 / 16-31)
    const int ntb  = (wave >> 1) * 8;   // 8 N tiles per wave -> covers 128
    const int lcol = lane & 15;
    const bf16* arow = h_lds + (size_t)(mt * 16 + lcol) * HH;

    for (int step = 0; step < SS; ++step) {
        const int t = dir ? (SS - 1 - step) : step;

        // ---- g_pre = h @ W_hh^T via WMMA, K-outer / 8 accumulators ----
        v8f acc[8] = {};
#pragma unroll 2
        for (int kk = 0; kk < HH; kk += 32) {
            v16bf a = load_a_frag(arow, kk, lane);   // from LDS, once per kk
#pragma unroll
            for (int jt = 0; jt < 8; ++jt) {
                const bf16* wrow = W + (size_t)((ntb + jt) * 16 + lcol) * HH;
                v16bf b = load_b_frag(wrow, kk, lane);
                acc[jt] = wmma_bf16(a, b, acc[jt]);
            }
        }
        {
            const int m0 = mt * 16 + ((lane < 16) ? 0 : 8);
#pragma unroll
            for (int jt = 0; jt < 8; ++jt) {
                const int ncol = (ntb + jt) * 16 + lcol;
#pragma unroll
                for (int i = 0; i < 8; ++i)
                    g_pre[(size_t)(m0 + i) * G4 + ncol] = acc[jt][i];
            }
        }
        __syncthreads();

        // ---- gates + state update (PyTorch order i,f,g,o) ----
        const float* xp_t = xp_d + (size_t)t * BB * G4;
#pragma unroll
        for (int q = 0; q < 16; ++q) {
            int p = q * 1024 + tid;
            int b = p >> 9, j = p & (HH - 1);
            const float* xr = xp_t + (size_t)b * G4;
            const float* gr = g_pre + (size_t)b * G4;
            float gi = gr[j]            + xr[j];
            float gf = gr[HH + j]       + xr[HH + j];
            float gg = gr[2 * HH + j]   + xr[2 * HH + j];
            float go = gr[3 * HH + j]   + xr[3 * HH + j];
            float c  = sigf(gf) * c_reg[q] + sigf(gi) * tanhf(gg);
            float h  = sigf(go) * tanhf(c);
            c_reg[q] = c;
            hs_d[(size_t)t * BB * HH + p] = h;
            h_lds[p] = (bf16)h;
        }
        __syncthreads();
    }
}

// ---------------- output projection: feats = [hf|hb] @ W_out^T + b_out ----
__global__ void __launch_bounds__(64)
feats_kernel(const float* __restrict__ hs, const float* __restrict__ W_out,
             const float* __restrict__ b_out, float* __restrict__ feats) {
    __shared__ float hbuf[2 * HH];
    const int m = blockIdx.x, tid = threadIdx.x;
    for (int k = tid; k < HH; k += 64) {
        hbuf[k]       = hs[(size_t)m * HH + k];                 // forward
        hbuf[HH + k]  = hs[(size_t)(MM + m) * HH + k];          // backward
    }
    __syncthreads();
    if (tid < TT) {
        float acc = b_out[tid];
        const float* w = W_out + (size_t)tid * 2 * HH;
#pragma unroll 8
        for (int k = 0; k < 2 * HH; ++k) acc = fmaf(hbuf[k], w[k], acc);
        feats[(size_t)m * TT + tid] = acc;
    }
}

// ---------------- Viterbi decode (one block per sentence) ----------------
__global__ void __launch_bounds__(64)
viterbi_kernel(const float* __restrict__ feats, const float* __restrict__ trans,
               int* __restrict__ bps, float* __restrict__ out) {
    __shared__ float fv[2][TT];
    __shared__ float tr[TT * TT];
    const int b = blockIdx.x, tid = threadIdx.x;
    for (int i = tid; i < TT * TT; i += 64) tr[i] = trans[i];
    if (tid < TT) fv[0][tid] = (tid == START_TAG) ? 0.0f : NEGV;
    __syncthreads();

    for (int s = 0; s < SS; ++s) {
        const int cur = s & 1, nxt = cur ^ 1;
        if (tid < TT) {
            float best = -3.4e38f; int bi = 0;
#pragma unroll
            for (int p = 0; p < TT; ++p) {
                float v = fv[cur][p] + tr[tid * TT + p];
                if (v > best) { best = v; bi = p; }
            }
            bps[((size_t)b * SS + s) * TT + tid] = bi;
            fv[nxt][tid] = best + feats[((size_t)s * BB + b) * TT + tid];
        }
        __syncthreads();
    }

    if (tid == 0) {
        const int fin = SS & 1;  // SS even -> final values in fv[0]
        float best = -3.4e38f; int last = 0;
        for (int t = 0; t < TT; ++t) {
            float v = fv[fin][t] + tr[END_TAG * TT + t];
            if (v > best) { best = v; last = t; }
        }
        float* row = out + (size_t)b * (SS + 2);
        row[0]      = (float)START_TAG;
        row[SS + 1] = (float)END_TAG;
        int tag = last;
        for (int s = SS - 1; s >= 0; --s) {
            row[1 + s] = (float)tag;
            tag = bps[((size_t)b * SS + s) * TT + tag];
        }
        out[(size_t)BB * (SS + 2) + b] = best;   // path_score
    }
}

// ---------------- launcher ----------------
extern "C" void kernel_launch(void* const* d_in, const int* in_sizes, int n_in,
                              void* d_out, int out_size, void* d_ws, size_t ws_size,
                              hipStream_t stream) {
    const int*   sent   = (const int*)d_in[0];
    const float* emb    = (const float*)d_in[1];
    const float* W_ih_f = (const float*)d_in[2];
    const float* W_hh_f = (const float*)d_in[3];
    const float* b_f    = (const float*)d_in[4];
    const float* W_ih_b = (const float*)d_in[5];
    const float* W_hh_b = (const float*)d_in[6];
    const float* b_b    = (const float*)d_in[7];
    const float* W_out  = (const float*)d_in[8];
    const float* b_out  = (const float*)d_in[9];
    const float* trans  = (const float*)d_in[10];
    const float* h0     = (const float*)d_in[11];
    const float* c0     = (const float*)d_in[12];
    float* out = (float*)d_out;

    char* ws = (char*)d_ws;
    bf16*  xbf   = (bf16*)(ws + OFF_X);
    bf16*  Wih   = (bf16*)(ws + OFF_WIH);
    bf16*  Whh   = (bf16*)(ws + OFF_WHH);
    float* xp    = (float*)(ws + OFF_XP);
    float* hs    = (float*)(ws + OFF_HS);
    float* feats = (float*)(ws + OFF_FEATS);
    int*   bps   = (int*)(ws + OFF_BPS);

    // weight casts (fp32 -> bf16, K padded for WMMA)
    cast_pad_kernel<<<(G4 * EP + 255) / 256, 256, 0, stream>>>(W_ih_f, Wih, G4, EE, EP);
    cast_pad_kernel<<<(G4 * EP + 255) / 256, 256, 0, stream>>>(W_ih_b, Wih + (size_t)G4 * EP, G4, EE, EP);
    cast_pad_kernel<<<(G4 * HH + 255) / 256, 256, 0, stream>>>(W_hh_f, Whh, G4, HH, HH);
    cast_pad_kernel<<<(G4 * HH + 255) / 256, 256, 0, stream>>>(W_hh_b, Whh + (size_t)G4 * HH, G4, HH, HH);

    // embedding gather + bf16 cast into [S,B,EP]
    embed_kernel<<<(MM * EP + 255) / 256, 256, 0, stream>>>(sent, emb, xbf);

    // input projection for both directions (WMMA bf16)
    dim3 gih(4, MM / 16, 2);
    ih_gemm_kernel<<<gih, 256, 0, stream>>>(xbf, Wih, b_f, b_b, xp);

    // persistent bidirectional recurrence: 288KB dynamic LDS per block
    const int lstm_lds = BB * G4 * 4 + BB * HH * 2;  // 294912 bytes
    hipFuncSetAttribute((const void*)lstm_kernel,
                        hipFuncAttributeMaxDynamicSharedMemorySize, lstm_lds);
    lstm_kernel<<<2, 1024, lstm_lds, stream>>>(xp, Whh, h0, c0, hs);

    // emission scores
    feats_kernel<<<MM, 64, 0, stream>>>(hs, W_out, b_out, feats);

    // CRF Viterbi decode + backtrace
    viterbi_kernel<<<BB, 64, 0, stream>>>(feats, trans, bps, out);
}
